// DGAD_net_11261404250140
// MI455X (gfx1250) — matrix operand order repaired and checked by reading, not compiled, not run on hardware
//
#include <hip/hip_runtime.h>

typedef __attribute__((ext_vector_type(16))) _Float16 v16h;
typedef __attribute__((ext_vector_type(8)))  _Float16 v8h;
typedef __attribute__((ext_vector_type(8)))  float    v8f;
typedef __attribute__((ext_vector_type(4)))  float    v4f;
typedef __attribute__((ext_vector_type(4)))  int      v4i;

#define LDS_RS 72   // padded LDS row stride in halves (144B = 36 dwords; conflict-friendly)

#if __has_builtin(__builtin_amdgcn_global_load_async_to_lds_b128)
#define USE_ASYNC_LDS 1
#define GAS1(p) ((__attribute__((address_space(1))) v4i*)(p))
#define LAS3(p) ((__attribute__((address_space(3))) v4i*)(p))
#else
#define USE_ASYNC_LDS 0
#endif

static __device__ __forceinline__ void wait_asynccnt0()
{
#if __has_builtin(__builtin_amdgcn_s_wait_asynccnt)
    __builtin_amdgcn_s_wait_asynccnt(0);
#else
    asm volatile("s_wait_asynccnt 0x0" ::: "memory");
#endif
}

// ---------------- mean-pool over trailing dim (one wave per row), f16 out ----------------
__global__ void pool_rows_mean(const float* __restrict__ X, _Float16* __restrict__ Y,
                               int rows, int hw, float inv)
{
    int wid  = (int)((blockIdx.x * blockDim.x + threadIdx.x) >> 5);
    int lane = threadIdx.x & 31;
    if (wid >= rows) return;
    const float* row = X + (size_t)wid * hw;
    float s = 0.f;
    if ((hw & 3) == 0) {
        int nv = hw >> 2;
        for (int i = lane; i < nv; i += 32) {
            v4f v = *(const v4f*)(row + 4 * i);
            s += v[0] + v[1] + v[2] + v[3];
        }
    } else {
        for (int i = lane; i < hw; i += 32) s += row[i];
    }
    for (int o = 16; o > 0; o >>= 1) s += __shfl_xor(s, o, 32);
    if (lane == 0) Y[wid] = (_Float16)(s * inv);
}

// ---------------- f32 -> f16 weight conversion ----------------
__global__ void cvt_f32_f16(const float* __restrict__ s, _Float16* __restrict__ d, int n)
{
    int i = blockIdx.x * blockDim.x + threadIdx.x;
    if (i < n) d[i] = (_Float16)s[i];
}

// ---------------- WMMA GEMM: Y[128,N] = A[128,K] @ W[N,K]^T, optional leaky_relu.
// Block = 256 threads = 8 waves; block owns one 64-wide N strip; wave w owns M tile w.
// B staged through double-buffered LDS (async global->LDS when available), one stage
// (64 K-halves) prefetched ahead; A prefetched one stage ahead into registers.
__global__ __launch_bounds__(256) void gemm_wmma(
    const _Float16* __restrict__ A,   // [128,K] f16
    const _Float16* __restrict__ W,   // [N,K]  f16
    float* __restrict__ Yf,           // [128,N] f32 (or null)
    _Float16* __restrict__ Yh,        // [128,N] f16 (or null)
    int N, int K, int do_lrelu)
{
    __shared__ _Float16 lds[2][64 * LDS_RS];
    const int tid  = threadIdx.x;
    const int lane = tid & 31;
    const int wave = tid >> 5;          // M tile 0..7
    const int ns   = blockIdx.x;        // N strip of 64
    const int h    = lane >> 4;
    const int ln   = lane & 15;
    const int nst  = K >> 6;            // stages of 64 K-halves

    // cooperative B stage mapping: thread -> (row 0..63, 16-half segment 0..3)
    const int brow = tid >> 2;
    const int bseg = (tid & 3) << 4;
    const _Float16* Bg = W + (size_t)(ns * 64 + brow) * K + bseg;
    const int       Bo = brow * LDS_RS + bseg;      // LDS offset in halves

    const _Float16* Arow = A + (size_t)(wave * 16 + ln) * K + 8 * h;

    v8h aR[4], aN[4];
#if !USE_ASYNC_LDS
    v8h bR0, bR1;
#endif

    // ---- prologue: stage 0 ----
#if USE_ASYNC_LDS
    __builtin_amdgcn_global_load_async_to_lds_b128(GAS1(Bg),     LAS3(&lds[0][Bo]),     0, 0);
    __builtin_amdgcn_global_load_async_to_lds_b128(GAS1(Bg + 8), LAS3(&lds[0][Bo + 8]), 0, 0);
#else
    bR0 = *(const v8h*)(Bg);
    bR1 = *(const v8h*)(Bg + 8);
#endif
    aR[0] = *(const v8h*)(Arow);
    aR[1] = *(const v8h*)(Arow + 16);
    aR[2] = *(const v8h*)(Arow + 32);
    aR[3] = *(const v8h*)(Arow + 48);
#if USE_ASYNC_LDS
    wait_asynccnt0();
#else
    *(v8h*)(&lds[0][Bo])     = bR0;
    *(v8h*)(&lds[0][Bo + 8]) = bR1;
#endif
    __syncthreads();

    v8f acc[4] = {};
    for (int s = 0; s < nst; ++s) {
        const int p = s & 1;
        // ---- prefetch stage s+1 ----
        if (s + 1 < nst) {
            const _Float16* Bg2 = Bg + (size_t)(s + 1) * 64;
#if USE_ASYNC_LDS
            __builtin_amdgcn_global_load_async_to_lds_b128(GAS1(Bg2),     LAS3(&lds[p ^ 1][Bo]),     0, 0);
            __builtin_amdgcn_global_load_async_to_lds_b128(GAS1(Bg2 + 8), LAS3(&lds[p ^ 1][Bo + 8]), 0, 0);
#else
            bR0 = *(const v8h*)(Bg2);
            bR1 = *(const v8h*)(Bg2 + 8);
#endif
            const _Float16* An = Arow + (size_t)(s + 1) * 64;
            aN[0] = *(const v8h*)(An);
            aN[1] = *(const v8h*)(An + 16);
            aN[2] = *(const v8h*)(An + 32);
            aN[3] = *(const v8h*)(An + 48);
        }
        // ---- consume stage s: 2 K-steps of 32, 4 N tiles each ----
        const _Float16* buf = &lds[p][0];
#pragma unroll
        for (int ks = 0; ks < 2; ++ks) {
            v16h af;
#pragma unroll
            for (int j = 0; j < 8; ++j) {
                af[j]     = aR[2 * ks][j];
                af[j + 8] = aR[2 * ks + 1][j];
            }
#pragma unroll
            for (int t = 0; t < 4; ++t) {
                const _Float16* bp = buf + (t * 16 + ln) * LDS_RS + ks * 32 + 8 * h;
                v8h b0 = *(const v8h*)bp;
                v8h b1 = *(const v8h*)(bp + 16);
                v16h bf;
#pragma unroll
                for (int j = 0; j < 8; ++j) { bf[j] = b0[j]; bf[j + 8] = b1[j]; }
                acc[t] = __builtin_amdgcn_wmma_f32_16x16x32_f16(
                    false, af, false, bf, (short)0, acc[t], false, false);
            }
        }
        // ---- commit prefetch ----
        if (s + 1 < nst) {
#if !USE_ASYNC_LDS
            *(v8h*)(&lds[p ^ 1][Bo])     = bR0;
            *(v8h*)(&lds[p ^ 1][Bo + 8]) = bR1;
#endif
            aR[0] = aN[0]; aR[1] = aN[1]; aR[2] = aN[2]; aR[3] = aN[3];
#if USE_ASYNC_LDS
            wait_asynccnt0();
#endif
        }
        __syncthreads();
    }
    // ---- epilogue: element i of lane -> row wave*16 + 8h + i, col ns*64 + t*16 + ln ----
#pragma unroll
    for (int t = 0; t < 4; ++t) {
#pragma unroll
        for (int i = 0; i < 8; ++i) {
            float v = acc[t][i];
            if (do_lrelu) v = v > 0.f ? v : 0.01f * v;
            size_t idx = (size_t)(wave * 16 + 8 * h + i) * N + (ns * 64 + t * 16 + ln);
            if (Yf) Yf[idx] = v;
            if (Yh) Yh[idx] = (_Float16)v;
        }
    }
}

// ---------------- concat builders (emit f16 for GEMM consumption) ----------------
__global__ void build_t_in(const float* __restrict__ shallow, const float* __restrict__ center,
                           _Float16* __restrict__ tin)
{
    int idx = blockIdx.x * blockDim.x + threadIdx.x;
    if (idx >= 128 * 64) return;
    int b = idx >> 6, j = idx & 63;
    float s = shallow[idx];
    tin[b * 128 + j]      = (_Float16)s;
    tin[b * 128 + 64 + j] = (_Float16)(s - center[j]);
}

__global__ void build_c_in(const float* __restrict__ origin, const float* __restrict__ proto,
                           const int* __restrict__ cat, _Float16* __restrict__ cin)
{
    int idx = blockIdx.x * blockDim.x + threadIdx.x;
    if (idx >= 128 * 64) return;
    int b = idx >> 6, j = idx & 63;
    float o = origin[idx];
    cin[b * 128 + j]      = (_Float16)o;
    cin[b * 128 + 64 + j] = (_Float16)(o - proto[cat[b] * 64 + j]);
}

// ---------------- prototype distances, argmax, cross-entropy ----------------
__global__ void sim_ce(const float* __restrict__ tex, const float* __restrict__ proto,
                       int* __restrict__ category, float* __restrict__ ce_out)
{
    __shared__ float red[128];
    int b = threadIdx.x;  // one block of 128
    float sim[4];
#pragma unroll
    for (int k = 0; k < 4; k++) {
        float s = 0.f;
        for (int j = 0; j < 64; j++) {
            float d = tex[b * 64 + j] - proto[k * 64 + j];
            s += d * d;
        }
        sim[k] = s;
    }
    int cat = 0; float mx = sim[0];
#pragma unroll
    for (int k = 1; k < 4; k++) if (sim[k] > mx) { mx = sim[k]; cat = k; }
    category[b] = cat;
    float se = 0.f;
#pragma unroll
    for (int k = 0; k < 4; k++) se += expf(sim[k] - mx);
    // -logp[argmax] = logsumexp(sim) - max(sim) = log(sum exp(sim - max))
    red[b] = logf(se);
    __syncthreads();
    for (int st = 64; st > 0; st >>= 1) { if (b < st) red[b] += red[b + st]; __syncthreads(); }
    if (b == 0) *ce_out = red[0] * (1.0f / 128.0f);
}

// ---------------- SVDD norms + final reductions ----------------
__global__ void finalize(const float* __restrict__ cf, const float* __restrict__ qf,
                         const float* __restrict__ center, const float* __restrict__ ce,
                         float* __restrict__ out)
{
    __shared__ float ro[128], rc[128], ra[128];
    int b = threadIdx.x;  // one block of 128
    float cs = 0.f, os = 0.f;
    for (int j = 0; j < 64; j++) {
        float dc = cf[b * 64 + j] - center[j]; cs += dc * dc;
        float dq = qf[b * 64 + j] - center[j]; os += dq * dq;
    }
    ro[b] = os; rc[b] = cs; ra[b] = fabsf(os - cs);
    __syncthreads();
    for (int st = 64; st > 0; st >>= 1) {
        if (b < st) { ro[b] += ro[b + st]; rc[b] += rc[b + st]; ra[b] += ra[b + st]; }
        __syncthreads();
    }
    if (b == 0) {
        out[0] = *ce;
        out[1] = ro[0] * (1.0f / 128.0f);
        out[2] = rc[0] * (1.0f / 128.0f);
        out[3] = ra[0] * (1.0f / 128.0f);
    }
}

extern "C" void kernel_launch(void* const* d_in, const int* in_sizes, int n_in,
                              void* d_out, int out_size, void* d_ws, size_t ws_size,
                              hipStream_t stream)
{
    (void)in_sizes; (void)n_in; (void)out_size; (void)ws_size;
    const float* x_mid     = (const float*)d_in[0];   // [128,512,28,28]
    const float* x_deep    = (const float*)d_in[1];   // [128,2048,7,7]
    const float* w_shallow = (const float*)d_in[2];   // [2048,512]
    const float* ow1 = (const float*)d_in[3];         // [1024,2048]
    const float* ow2 = (const float*)d_in[4];         // [512,1024]
    const float* ow3 = (const float*)d_in[5];         // [64,512]
    const float* sw1 = (const float*)d_in[6];
    const float* sw2 = (const float*)d_in[7];
    const float* sw3 = (const float*)d_in[8];
    const float* tw1 = (const float*)d_in[9];         // [64,128]
    const float* tw2 = (const float*)d_in[10];        // [64,64]
    const float* cw1 = (const float*)d_in[11];
    const float* cw2 = (const float*)d_in[12];
    const float* qw1 = (const float*)d_in[13];
    const float* qw2 = (const float*)d_in[14];
    const float* center = (const float*)d_in[15];     // [64]
    const float* proto  = (const float*)d_in[16];     // [4,64]
    float* out = (float*)d_out;                       // 4 floats

    // ---- workspace carve-out ----
    size_t off = 0;
    auto alloc = [&](size_t bytes) -> void* {
        void* p = (char*)d_ws + off;
        off += (bytes + 255) & ~(size_t)255;
        return p;
    };
    _Float16* wshal_h = (_Float16*)alloc((size_t)2048 * 512 * 2);
    _Float16* ow1h = (_Float16*)alloc((size_t)1024 * 2048 * 2);
    _Float16* ow2h = (_Float16*)alloc((size_t)512 * 1024 * 2);
    _Float16* ow3h = (_Float16*)alloc((size_t)64 * 512 * 2);
    _Float16* sw1h = (_Float16*)alloc((size_t)1024 * 2048 * 2);
    _Float16* sw2h = (_Float16*)alloc((size_t)512 * 1024 * 2);
    _Float16* sw3h = (_Float16*)alloc((size_t)64 * 512 * 2);
    _Float16* tw1h = (_Float16*)alloc((size_t)64 * 128 * 2);
    _Float16* tw2h = (_Float16*)alloc((size_t)64 * 64 * 2);
    _Float16* cw1h = (_Float16*)alloc((size_t)64 * 128 * 2);
    _Float16* cw2h = (_Float16*)alloc((size_t)64 * 64 * 2);
    _Float16* qw1h = (_Float16*)alloc((size_t)64 * 64 * 2);
    _Float16* qw2h = (_Float16*)alloc((size_t)64 * 64 * 2);
    _Float16* xm_pool = (_Float16*)alloc((size_t)128 * 512 * 2);
    _Float16* xd_pool = (_Float16*)alloc((size_t)128 * 2048 * 2);
    _Float16* s0h     = (_Float16*)alloc((size_t)128 * 2048 * 2);
    _Float16* h1h     = (_Float16*)alloc((size_t)128 * 1024 * 2);
    _Float16* h2h     = (_Float16*)alloc((size_t)128 * 512 * 2);
    float*    shallow = (float*)alloc((size_t)128 * 64 * 4);
    _Float16* o1h     = (_Float16*)alloc((size_t)128 * 1024 * 2);
    _Float16* o2h     = (_Float16*)alloc((size_t)128 * 512 * 2);
    float*    originF = (float*)alloc((size_t)128 * 64 * 4);
    _Float16* originH = (_Float16*)alloc((size_t)128 * 64 * 2);
    _Float16* t_inH   = (_Float16*)alloc((size_t)128 * 128 * 2);
    _Float16* t1h     = (_Float16*)alloc((size_t)128 * 64 * 2);
    float*    texture = (float*)alloc((size_t)128 * 64 * 4);
    _Float16* c_inH   = (_Float16*)alloc((size_t)128 * 128 * 2);
    _Float16* c1h     = (_Float16*)alloc((size_t)128 * 64 * 2);
    float*    cfF     = (float*)alloc((size_t)128 * 64 * 4);
    _Float16* q1h     = (_Float16*)alloc((size_t)128 * 64 * 2);
    float*    qfF     = (float*)alloc((size_t)128 * 64 * 4);
    int*      categ   = (int*)alloc((size_t)128 * 4);
    float*    ce      = (float*)alloc(256);

    auto cvt = [&](const float* s, _Float16* d, int n) {
        cvt_f32_f16<<<(n + 255) / 256, 256, 0, stream>>>(s, d, n);
    };
    auto gemm = [&](const _Float16* A, const _Float16* Wh, float* Yf, _Float16* Yh,
                    int N, int K, int lrelu) {
        gemm_wmma<<<N / 64, 256, 0, stream>>>(A, Wh, Yf, Yh, N, K, lrelu);
    };

    // 1) weights -> f16 (once per call; ~13 MB, trivial vs 256 MB pooling traffic)
    cvt(w_shallow, wshal_h, 2048 * 512);
    cvt(ow1, ow1h, 1024 * 2048); cvt(ow2, ow2h, 512 * 1024); cvt(ow3, ow3h, 64 * 512);
    cvt(sw1, sw1h, 1024 * 2048); cvt(sw2, sw2h, 512 * 1024); cvt(sw3, sw3h, 64 * 512);
    cvt(tw1, tw1h, 64 * 128); cvt(tw2, tw2h, 64 * 64);
    cvt(cw1, cw1h, 64 * 128); cvt(cw2, cw2h, 64 * 64);
    cvt(qw1, qw1h, 64 * 64);  cvt(qw2, qw2h, 64 * 64);

    // 2) pooling (bandwidth-bound roofline term: 256 MB -> ~11 us at 23.3 TB/s)
    {
        int rows = 128 * 512;   // x_mid rows of 784
        pool_rows_mean<<<(rows + 7) / 8, 256, 0, stream>>>(x_mid, xm_pool, rows, 784, 1.0f / 784.0f);
        rows = 128 * 2048;      // x_deep rows of 49
        pool_rows_mean<<<(rows + 7) / 8, 256, 0, stream>>>(x_deep, xd_pool, rows, 49, 1.0f / 49.0f);
    }

    // 3) pooled 1x1-conv (mean commutes with linear conv), no activation
    gemm(xm_pool, wshal_h, nullptr, s0h, 2048, 512, 0);
    // 4) shallow MLP
    gemm(s0h, sw1h, nullptr, h1h, 1024, 2048, 1);
    gemm(h1h, sw2h, nullptr, h2h, 512, 1024, 1);
    gemm(h2h, sw3h, shallow, nullptr, 64, 512, 1);
    // 5) origin MLP
    gemm(xd_pool, ow1h, nullptr, o1h, 1024, 2048, 1);
    gemm(o1h, ow2h, nullptr, o2h, 512, 1024, 1);
    gemm(o2h, ow3h, originF, originH, 64, 512, 1);
    // 6) texture head
    build_t_in<<<(128 * 64 + 255) / 256, 256, 0, stream>>>(shallow, center, t_inH);
    gemm(t_inH, tw1h, nullptr, t1h, 64, 128, 1);
    gemm(t1h, tw2h, texture, nullptr, 64, 64, 1);
    // 7) prototype distances, argmax category, CE
    sim_ce<<<1, 128, 0, stream>>>(texture, proto, categ, ce);
    // 8) class head (gathers per-sample prototype)
    build_c_in<<<(128 * 64 + 255) / 256, 256, 0, stream>>>(originF, proto, categ, c_inH);
    gemm(c_inH, cw1h, nullptr, c1h, 64, 128, 1);
    gemm(c1h, cw2h, cfF, nullptr, 64, 64, 1);
    // 9) q head
    gemm(originH, qw1h, nullptr, q1h, 64, 64, 1);
    gemm(q1h, qw2h, qfF, nullptr, 64, 64, 1);
    // 10) SVDD norms, align, means -> 4 outputs
    finalize<<<1, 128, 0, stream>>>(cfF, qfF, center, ce, out);
}
